// MoE_37005438223072
// MI455X (gfx1250) — matrix operand order repaired and checked by reading, not compiled, not run on hardware
//
#include <hip/hip_runtime.h>

#define Bb 4
#define Tt 2048
#define Dd 1024
#define Ee 16
#define Hh 4096
#define Cc 160
#define Mm (Bb*Cc)   // 640 rows per expert

typedef __attribute__((ext_vector_type(16))) __bf16 v16bf;
typedef __attribute__((ext_vector_type(8)))  __bf16 v8bf;
typedef __attribute__((ext_vector_type(8)))  float  v8f;

// ---------------------------------------------------------------------------
// Transpose + fp32 -> bf16 convert:  out[e][c][r] = bf16(in[e][r][c])
// in: [E, R, C], out: [E, C, R].  grid (C/32, R/32, E), block (32, 8)
// ---------------------------------------------------------------------------
__global__ __launch_bounds__(256)
void transpose_conv_bf16(const float* __restrict__ in, __bf16* __restrict__ out,
                         int R, int Ccols) {
    __shared__ float tile[32][33];
    const int e  = blockIdx.z;
    const size_t plane = (size_t)R * Ccols;
    const float* src = in  + (size_t)e * plane;
    __bf16*      dst = out + (size_t)e * plane;
    const int c0 = blockIdx.x * 32;
    const int r0 = blockIdx.y * 32;
    const int tx = threadIdx.x;
    const int ty = threadIdx.y;
    #pragma unroll
    for (int i = ty; i < 32; i += 8)
        tile[i][tx] = src[(size_t)(r0 + i) * Ccols + (c0 + tx)];
    __syncthreads();
    #pragma unroll
    for (int i = ty; i < 32; i += 8)
        dst[(size_t)(c0 + i) * R + (r0 + tx)] = (__bf16)tile[tx][i];
}

// ---------------------------------------------------------------------------
// Gating: one wave32 per token. logits = x[token,:] @ wg, softmax, stable
// descending sort, dynamic-K (cum < 0.8, slot 0 forced), renormalize.
// Writes probs[B*T,E] (softmax) and gatew[B*T,E] (selected renorm weight or 0)
// ---------------------------------------------------------------------------
__global__ __launch_bounds__(256)
void gating_kernel(const float* __restrict__ x, const float* __restrict__ wg,
                   float* __restrict__ probs, float* __restrict__ gatew) {
    const int token = blockIdx.x * 8 + (threadIdx.x >> 5);
    const int lane  = threadIdx.x & 31;
    const float* xr = x + (size_t)token * Dd;

    float acc[Ee];
    #pragma unroll
    for (int e = 0; e < Ee; e++) acc[e] = 0.f;

    for (int d = lane; d < Dd; d += 32) {
        const float xv = xr[d];
        const float* wr = wg + (size_t)d * Ee;
        #pragma unroll
        for (int e = 0; e < Ee; e++) acc[e] += xv * wr[e];
    }
    #pragma unroll
    for (int off = 16; off > 0; off >>= 1) {
        #pragma unroll
        for (int e = 0; e < Ee; e++) acc[e] += __shfl_down(acc[e], off, 32);
    }

    if (lane == 0) {
        float mx = acc[0];
        #pragma unroll
        for (int e = 1; e < Ee; e++) mx = fmaxf(mx, acc[e]);
        float p[Ee]; float sum = 0.f;
        #pragma unroll
        for (int e = 0; e < Ee; e++) { p[e] = __expf(acc[e] - mx); sum += p[e]; }
        const float inv = 1.f / sum;
        #pragma unroll
        for (int e = 0; e < Ee; e++) p[e] *= inv;

        // stable insertion sort, descending
        float ps[Ee]; int order[Ee];
        for (int e = 0; e < Ee; e++) { ps[e] = p[e]; order[e] = e; }
        for (int i = 1; i < Ee; i++) {
            float v = ps[i]; int oi = order[i]; int j = i - 1;
            while (j >= 0 && ps[j] < v) { ps[j+1] = ps[j]; order[j+1] = order[j]; j--; }
            ps[j+1] = v; order[j+1] = oi;
        }
        // dynamic K: keep while inclusive cumsum < 0.8 (slot 0 always kept)
        float csum = 0.f; int k = 0;
        for (int i = 0; i < Ee; i++) {
            csum += ps[i];
            if (i == 0 || csum < 0.8f) k++;
            else break;
        }
        float ssel = 0.f;
        for (int i = 0; i < k; i++) ssel += ps[i];
        const float invs = 1.f / fmaxf(ssel, 1e-9f);

        float gw[Ee];
        #pragma unroll
        for (int e = 0; e < Ee; e++) gw[e] = 0.f;
        for (int i = 0; i < k; i++) gw[order[i]] = ps[i] * invs;

        const size_t off = (size_t)token * Ee;
        #pragma unroll
        for (int e = 0; e < Ee; e++) { probs[off + e] = p[e]; gatew[off + e] = gw[e]; }
    }
}

// ---------------------------------------------------------------------------
// Capacity: one block per (b, e). Exclusive scan of mask over the T tokens
// (deterministic token order). slot_idx[b,t,e] = position (or -1 if dropped /
// not selected). token_list[(e*B+b)*C + pos] = t.
// ---------------------------------------------------------------------------
__global__ __launch_bounds__(256)
void capacity_kernel(const float* __restrict__ gatew,
                     int* __restrict__ slot_idx, int* __restrict__ token_list) {
    const int b = blockIdx.x / Ee;
    const int e = blockIdx.x % Ee;
    const int tid = threadIdx.x;
    __shared__ int ssum[256];

    int* tl = token_list + ((size_t)e * Bb + b) * Cc;
    for (int c = tid; c < Cc; c += 256) tl[c] = -1;
    __syncthreads();

    const int tbase = tid * 8;
    int m[8]; int cnt = 0;
    #pragma unroll
    for (int i = 0; i < 8; i++) {
        const size_t idx = ((size_t)(b * Tt) + tbase + i) * Ee + e;
        m[i] = (gatew[idx] > 0.f) ? 1 : 0;
        cnt += m[i];
    }
    ssum[tid] = cnt;
    __syncthreads();
    for (int off = 1; off < 256; off <<= 1) {
        int v = (tid >= off) ? ssum[tid - off] : 0;
        __syncthreads();
        ssum[tid] += v;
        __syncthreads();
    }
    int pos = ssum[tid] - cnt;  // exclusive prefix for this thread
    #pragma unroll
    for (int i = 0; i < 8; i++) {
        const int t = tbase + i;
        const size_t idx = ((size_t)(b * Tt) + t) * Ee + e;
        int s = -1;
        if (m[i]) {
            if (pos < Cc) { s = pos; tl[pos] = t; }
            pos++;
        }
        slot_idx[idx] = s;
    }
}

// ---------------------------------------------------------------------------
// Gather: Xe[(e*B+b)*C + c][:] = bf16(x[b, token, :]) or zeros. One block/row.
// ---------------------------------------------------------------------------
__global__ __launch_bounds__(256)
void gather_kernel(const float* __restrict__ x, const int* __restrict__ token_list,
                   __bf16* __restrict__ Xe) {
    const int row = blockIdx.x;                 // (e*B + b)*C + c
    const int b = (row / Cc) % Bb;
    const int t = token_list[row];
    __bf16* dst = Xe + (size_t)row * Dd;
    const int tid = threadIdx.x;
    if (t < 0) {
        #pragma unroll
        for (int i = 0; i < 4; i++) dst[tid + i * 256] = (__bf16)0.0f;
    } else {
        const float* src = x + ((size_t)b * Tt + t) * Dd;
        #pragma unroll
        for (int i = 0; i < 4; i++) dst[tid + i * 256] = (__bf16)src[tid + i * 256];
    }
}

// ---------------------------------------------------------------------------
// WMMA bf16 GEMM:  D[e] = act(A[e] @ B[e])
//   A:  [E, M, K] bf16 row-major
//   Bt: [E, N, K] bf16 (B transposed -> per-lane fragment loads contiguous)
//   RELU_BF16: relu + bf16 output (layer 1), else fp32 output (layer 2)
// Block = 256 threads = 8 waves (2 in M x 4 in N); wave tile 32x64 -> 8 accs.
// A/B fragments (16-bit A 16x32, B 32x16) both use contiguous k = 16*h + s; f32 C/D per CDNA5 ISA 7.12.2.
// ---------------------------------------------------------------------------
__device__ inline v16bf load_a_frag(const __bf16* __restrict__ row, int k0, int h) {
    v8bf lo = *(const v8bf*)(row + k0 + 16 * h);
    v8bf hi = *(const v8bf*)(row + k0 + 16 * h + 8);
    v16bf r;
    #pragma unroll
    for (int i = 0; i < 8; i++) { r[i] = lo[i]; r[i + 8] = hi[i]; }
    return r;
}

template <bool RELU_BF16>
__global__ __launch_bounds__(256)
void moe_gemm(const __bf16* __restrict__ Aall, const __bf16* __restrict__ Btall,
              void* __restrict__ Dall, int M, int N, int K) {
    const int e = blockIdx.z;
    const __bf16* A  = Aall  + (size_t)e * M * K;
    const __bf16* Bt = Btall + (size_t)e * N * K;

    const int tid  = threadIdx.x;
    const int lane = tid & 31;
    const int wave = tid >> 5;
    const int wn = wave & 3;      // 4 waves across N, 64 cols each
    const int wm = wave >> 2;     // 2 waves across M, 32 rows each
    const int n_base = blockIdx.x * 256 + wn * 64;
    const int m_base = blockIdx.y * 64 + wm * 32;
    const int h   = lane >> 4;    // lane half
    const int l15 = lane & 15;

    v8f acc[2][4];
    #pragma unroll
    for (int i = 0; i < 2; i++)
        #pragma unroll
        for (int j = 0; j < 4; j++)
            #pragma unroll
            for (int q = 0; q < 8; q++) acc[i][j][q] = 0.f;

    const __bf16* arow0 = A + (size_t)(m_base + l15) * K;
    const __bf16* arow1 = A + (size_t)(m_base + 16 + l15) * K;
    const __bf16* bcol0 = Bt + (size_t)(n_base +  0 + l15) * K + 16 * h;
    const __bf16* bcol1 = Bt + (size_t)(n_base + 16 + l15) * K + 16 * h;
    const __bf16* bcol2 = Bt + (size_t)(n_base + 32 + l15) * K + 16 * h;
    const __bf16* bcol3 = Bt + (size_t)(n_base + 48 + l15) * K + 16 * h;

    for (int k0 = 0; k0 < K; k0 += 32) {
        const v16bf a0 = load_a_frag(arow0, k0, h);
        const v16bf a1 = load_a_frag(arow1, k0, h);
        const v16bf b0 = *(const v16bf*)(bcol0 + k0);
        const v16bf b1 = *(const v16bf*)(bcol1 + k0);
        const v16bf b2 = *(const v16bf*)(bcol2 + k0);
        const v16bf b3 = *(const v16bf*)(bcol3 + k0);
        acc[0][0] = __builtin_amdgcn_wmma_f32_16x16x32_bf16(false, a0, false, b0, (short)0, acc[0][0], false, false);
        acc[1][0] = __builtin_amdgcn_wmma_f32_16x16x32_bf16(false, a1, false, b0, (short)0, acc[1][0], false, false);
        acc[0][1] = __builtin_amdgcn_wmma_f32_16x16x32_bf16(false, a0, false, b1, (short)0, acc[0][1], false, false);
        acc[1][1] = __builtin_amdgcn_wmma_f32_16x16x32_bf16(false, a1, false, b1, (short)0, acc[1][1], false, false);
        acc[0][2] = __builtin_amdgcn_wmma_f32_16x16x32_bf16(false, a0, false, b2, (short)0, acc[0][2], false, false);
        acc[1][2] = __builtin_amdgcn_wmma_f32_16x16x32_bf16(false, a1, false, b2, (short)0, acc[1][2], false, false);
        acc[0][3] = __builtin_amdgcn_wmma_f32_16x16x32_bf16(false, a0, false, b3, (short)0, acc[0][3], false, false);
        acc[1][3] = __builtin_amdgcn_wmma_f32_16x16x32_bf16(false, a1, false, b3, (short)0, acc[1][3], false, false);
    }

    // D layout: VGPR r, lane half h -> M = m_tile + 8*h + r, N = n_tile + l15
    #pragma unroll
    for (int i = 0; i < 2; i++) {
        #pragma unroll
        for (int j = 0; j < 4; j++) {
            const int n = n_base + j * 16 + l15;
            #pragma unroll
            for (int r = 0; r < 8; r++) {
                const int m = m_base + i * 16 + 8 * h + r;
                float v = acc[i][j][r];
                const size_t idx = ((size_t)e * M + m) * N + n;
                if (RELU_BF16) {
                    v = v > 0.f ? v : 0.f;
                    ((__bf16*)Dall)[idx] = (__bf16)v;
                } else {
                    ((float*)Dall)[idx] = v;
                }
            }
        }
    }
}

// ---------------------------------------------------------------------------
// Combine: out[b,t,:] = sum_e gatew[b,t,e] * Oute[e, b*C+slot, :]
// Fixed expert-order accumulation -> deterministic. One block per token.
// ---------------------------------------------------------------------------
__global__ __launch_bounds__(256)
void combine_kernel(const float* __restrict__ Oute, const int* __restrict__ slot_idx,
                    const float* __restrict__ gatew, float* __restrict__ out) {
    const int token = blockIdx.x;        // b*T + t
    const int b = token / Tt;
    const int tid = threadIdx.x;
    float acc[4] = {0.f, 0.f, 0.f, 0.f};
    const size_t goff = (size_t)token * Ee;
    for (int e = 0; e < Ee; e++) {
        const int s = slot_idx[goff + e];
        if (s < 0) continue;
        const float w = gatew[goff + e];
        const float* src = Oute + (((size_t)e * Bb + b) * Cc + s) * Dd;
        #pragma unroll
        for (int i = 0; i < 4; i++) acc[i] += w * src[tid + i * 256];
    }
    float* dst = out + (size_t)token * Dd;
    #pragma unroll
    for (int i = 0; i < 4; i++) dst[tid + i * 256] = acc[i];
}

// ---------------------------------------------------------------------------
// Aux loss: mean_{b,e}( density * density_proxy ) * E^2 * 0.01
// Single block; 4 threads per (b,e) pair.
// ---------------------------------------------------------------------------
__global__ __launch_bounds__(256)
void aux_kernel(const float* __restrict__ probs, const float* __restrict__ gatew,
                float* __restrict__ out_aux) {
    __shared__ float smask[256], sprob[256];
    const int tid = threadIdx.x;
    const int pair = tid >> 2;          // 0..63
    const int sub  = tid & 3;
    const int b = pair / Ee;
    const int e = pair % Ee;
    float sm = 0.f, sp = 0.f;
    for (int t = sub; t < Tt; t += 4) {
        const size_t idx = ((size_t)(b * Tt) + t) * Ee + e;
        sm += (gatew[idx] > 0.f) ? 1.f : 0.f;
        sp += probs[idx];
    }
    smask[tid] = sm; sprob[tid] = sp;
    __syncthreads();
    if (sub == 0) {
        const float dm = (smask[tid] + smask[tid+1] + smask[tid+2] + smask[tid+3]) / (float)Tt;
        const float dp = (sprob[tid] + sprob[tid+1] + sprob[tid+2] + sprob[tid+3]) / (float)Tt;
        smask[tid] = dm * dp;
    }
    __syncthreads();
    if (tid == 0) {
        float s = 0.f;
        for (int q = 0; q < 64; q++) s += smask[q * 4];
        out_aux[0] = (s / (float)(Bb * Ee)) * (float)(Ee * Ee) * 0.01f;
    }
}

// ---------------------------------------------------------------------------
extern "C" void kernel_launch(void* const* d_in, const int* in_sizes, int n_in,
                              void* d_out, int out_size, void* d_ws, size_t ws_size,
                              hipStream_t stream) {
    (void)in_sizes; (void)n_in; (void)out_size; (void)ws_size;
    const float* x  = (const float*)d_in[0];   // [B,T,D]
    const float* wg = (const float*)d_in[1];   // [D,E]
    const float* w1 = (const float*)d_in[2];   // [E,D,H]
    const float* w2 = (const float*)d_in[3];   // [E,H,D]
    float* out = (float*)d_out;                // [B,T,D] + 1 (aux)

    char* p = (char*)d_ws;
    auto carve = [&](size_t bytes) -> void* {
        void* r = (void*)p;
        p += (bytes + 255) & ~(size_t)255;
        return r;
    };
    float*  probs = (float*) carve((size_t)Bb * Tt * Ee * 4);
    float*  gatew = (float*) carve((size_t)Bb * Tt * Ee * 4);
    int*    slot  = (int*)   carve((size_t)Bb * Tt * Ee * 4);
    int*    tlist = (int*)   carve((size_t)Ee * Bb * Cc * 4);
    __bf16* Xe    = (__bf16*)carve((size_t)Ee * Mm * Dd * 2);
    __bf16* W1t   = (__bf16*)carve((size_t)Ee * Dd * Hh * 2);
    __bf16* W2t   = (__bf16*)carve((size_t)Ee * Hh * Dd * 2);
    __bf16* Hid   = (__bf16*)carve((size_t)Ee * Mm * Hh * 2);
    float*  Oute  = (float*) carve((size_t)Ee * Mm * Dd * 4);

    // Weight convert + transpose to [N,K] bf16
    transpose_conv_bf16<<<dim3(Hh/32, Dd/32, Ee), dim3(32, 8), 0, stream>>>(w1, W1t, Dd, Hh);
    transpose_conv_bf16<<<dim3(Dd/32, Hh/32, Ee), dim3(32, 8), 0, stream>>>(w2, W2t, Hh, Dd);

    // Gating + capacity + gather
    gating_kernel<<<dim3((Bb * Tt) / 8), dim3(256), 0, stream>>>(x, wg, probs, gatew);
    capacity_kernel<<<dim3(Bb * Ee), dim3(256), 0, stream>>>(gatew, slot, tlist);
    gather_kernel<<<dim3(Ee * Bb * Cc), dim3(256), 0, stream>>>(x, tlist, Xe);

    // Expert FFN via WMMA bf16
    moe_gemm<true ><<<dim3(Hh/256, Mm/64, Ee), dim3(256), 0, stream>>>(Xe,  W1t, (void*)Hid,  Mm, Hh, Dd);
    moe_gemm<false><<<dim3(Dd/256, Mm/64, Ee), dim3(256), 0, stream>>>(Hid, W2t, (void*)Oute, Mm, Dd, Hh);

    // Combine + aux loss
    combine_kernel<<<dim3(Bb * Tt), dim3(256), 0, stream>>>(Oute, slot, gatew, out);
    aux_kernel<<<dim3(1), dim3(256), 0, stream>>>(probs, gatew, out + (size_t)Bb * Tt * Dd);
}
